// RPTAttention_63797444215057
// MI455X (gfx1250) — matrix operand order repaired and checked by reading, not compiled, not run on hardware
//
#include <hip/hip_runtime.h>
#include <hip/hip_bf16.h>

typedef unsigned short u16;
typedef unsigned int   u32;
typedef unsigned long long u64;

typedef __attribute__((ext_vector_type(16))) __bf16 v16bf;
typedef __attribute__((ext_vector_type(8)))  float  v8f;
typedef __attribute__((ext_vector_type(4)))  u32    v4u;
typedef __attribute__((ext_vector_type(8)))  int    v8i;
typedef __attribute__((ext_vector_type(4)))  int    v4i;

union Frag {
    v16bf v;
    v4u   q[2];
    u32   u[8];
    u16   h[16];
};

#ifdef __has_builtin
#if __has_builtin(__builtin_amdgcn_tensor_load_to_lds) && \
    __has_builtin(__builtin_amdgcn_s_wait_tensorcnt)
#define USE_TDM 1
#endif
#endif
#ifndef USE_TDM
#define USE_TDM 0
#endif

__device__ __forceinline__ u16 f2bf(float f) {
    u32 u = __float_as_uint(f);
    u32 r = u + 0x7FFFu + ((u >> 16) & 1u);   // round-to-nearest-even
    return (u16)(r >> 16);
}

__device__ __forceinline__ v8f wmma_bf16(v16bf a, v16bf b, v8f c) {
    return __builtin_amdgcn_wmma_f32_16x16x32_bf16(
        false, a, false, b, (short)0, c, false, false);
}

#if USE_TDM
// Issue a TDM load of a 2-D bf16 tile (tile_dim0=32 elems contiguous,
// tile_dim1=128 rows, row stride = K elems) into LDS at lds_off.
__device__ __forceinline__ void tdm_load_tile32x128(u32 lds_off, const u16* gptr,
                                                    int K, int N) {
    u64 ga = (u64)(const void*)gptr;
    v4u g0;
    g0[0] = 1u;                                   // count=1, user descriptor
    g0[1] = lds_off;                              // lds_addr (bytes)
    g0[2] = (u32)(ga & 0xFFFFFFFFull);            // global_addr[31:0]
    g0[3] = (u32)((ga >> 32) & 0x01FFFFFFull) | (2u << 30);  // addr[56:32]|type=2

    u32 td0 = (u32)K, td1 = (u32)N;
    u64 st0 = (u64)K;                             // tensor_dim0_stride (elems)
    v8i g1;
    g1[0] = (int)(1u << 16);                      // data_size=1 (2 bytes)
    g1[1] = (int)((td0 & 0xFFFFu) << 16);         // tensor_dim0[15:0]
    g1[2] = (int)(((td0 >> 16) & 0xFFFFu) | ((td1 & 0xFFFFu) << 16));
    g1[3] = (int)(((td1 >> 16) & 0xFFFFu) | (32u << 16));    // tile_dim0=32
    g1[4] = (int)128u;                            // tile_dim1=128, tile_dim2=0
    g1[5] = (int)(st0 & 0xFFFFFFFFull);
    g1[6] = (int)((st0 >> 32) & 0xFFFFull);
    g1[7] = 0;

    v4i z4 = {0, 0, 0, 0};
#if __clang_major__ >= 23
    v8i z8 = {0, 0, 0, 0, 0, 0, 0, 0};
    __builtin_amdgcn_tensor_load_to_lds(g0, g1, z4, z4, z8, 0);
#else
    __builtin_amdgcn_tensor_load_to_lds(g0, g1, z4, z4, 0);
#endif
}
#endif

// ---------------------------------------------------------------------------
// fp32 -> bf16 convert (row-major copy)
// ---------------------------------------------------------------------------
__global__ __launch_bounds__(256)
void cvt_bf16_kernel(const float* __restrict__ in, u16* __restrict__ out, int n) {
    int i = blockIdx.x * blockDim.x + threadIdx.x;
    if (i < n) out[i] = f2bf(in[i]);
}

// ---------------------------------------------------------------------------
// Weight transpose: fp32 W[K][N] -> bf16 WT[N][K] (contiguous writes)
// ---------------------------------------------------------------------------
__global__ __launch_bounds__(256)
void wtrans_kernel(const float* __restrict__ in, u16* __restrict__ out,
                   int K, int N) {
    int idx = blockIdx.x * blockDim.x + threadIdx.x;
    if (idx >= K * N) return;
    int k = idx % K;
    int n = idx / K;
    out[idx] = f2bf(in[(size_t)k * N + n]);   // out[n*K + k]
}

// ---------------------------------------------------------------------------
// RoPE: fp32 (S, nh, 128) -> bf16 (S, nh, 128)
// ---------------------------------------------------------------------------
__global__ __launch_bounds__(256)
void rope_kernel(const float* __restrict__ in, u16* __restrict__ out, int nh) {
    int idx = blockIdx.x * blockDim.x + threadIdx.x;
    int total = 2048 * nh * 64;
    if (idx >= total) return;
    int j  = idx % 64;
    int t  = idx / 64;
    int hh = t % nh;
    int s  = t / nh;
    float freq = __powf(10000.0f, -(2.0f * (float)j) / 128.0f);
    float ang  = (float)s * freq;
    float c = __cosf(ang), sn = __sinf(ang);
    size_t base = (((size_t)s * nh + hh) * 128) + 2 * j;
    float re = in[base], im = in[base + 1];
    out[base]     = f2bf(re * c - im * sn);
    out[base + 1] = f2bf(re * sn + im * c);
}

// ---------------------------------------------------------------------------
// V transpose: fp32 (S, 4, 128) -> bf16 Vt[hkv][d][s]
// ---------------------------------------------------------------------------
__global__ __launch_bounds__(256)
void vtrans_kernel(const float* __restrict__ Vf, u16* __restrict__ Vt) {
    int idx = blockIdx.x * blockDim.x + threadIdx.x;
    if (idx >= 2048 * 4 * 128) return;
    int d  = idx % 128;
    int t  = idx / 128;
    int hh = t % 4;
    int s  = t / 4;
    Vt[((size_t)hh * 128 + d) * 2048 + s] = f2bf(Vf[idx]);
}

// ---------------------------------------------------------------------------
// bf16 GEMM: C(f32, MxN) = A(bf16, MxK) @ B, with B given pre-transposed
// as BT[N][K] bf16. 256 threads = 8 waves; block tile 128(M) x 128(N).
// A streams from L2 (no cross-wave reuse); BT tile (8x wave reuse) staged
// in LDS via TDM (double buffered) and read as ds_load_b128 fragments.
// ---------------------------------------------------------------------------
__global__ __launch_bounds__(256)
void gemm_bf16_kernel(const u16* __restrict__ A, const u16* __restrict__ BT,
                      float* __restrict__ C, int M, int N, int K) {
#if USE_TDM
    __shared__ u16 Btl[2][128 * 32];   // [buf][n][k]
#else
    __shared__ u16 Btl[1][128 * 32];
#endif

    int tid  = threadIdx.x;
    int wave = tid >> 5;
    int lane = tid & 31;
    int lane16 = lane & 15;
    int hi     = lane >> 4;

    int m0 = blockIdx.y * 128;
    int n0 = blockIdx.x * 128;

    v8f acc[8];
#pragma unroll
    for (int nb = 0; nb < 8; ++nb) acc[nb] = (v8f){0,0,0,0,0,0,0,0};

    const u32* Ag = (const u32*)(A + (size_t)(m0 + wave * 16 + lane16) * K);

#if USE_TDM
    u32 lds0 = (u32)(u64)(const void*)&Btl[0][0];
    u32 lds1 = (u32)(u64)(const void*)&Btl[1][0];
    if (wave == 0)
        tdm_load_tile32x128(lds0, BT + (size_t)n0 * K, K, N);
#endif

    for (int kk = 0; kk < K; kk += 32) {
#if USE_TDM
        int buf = (kk >> 5) & 1;
        if (wave == 0) {
            if (kk + 32 < K) {
                tdm_load_tile32x128(buf ? lds0 : lds1,
                                    BT + (size_t)n0 * K + kk + 32, K, N);
                __builtin_amdgcn_s_wait_tensorcnt((short)1);
            } else {
                __builtin_amdgcn_s_wait_tensorcnt((short)0);
            }
        }
        __syncthreads();
        const u16* Bt = &Btl[buf][0];
#else
        // cooperative stage: 2048 dwords, contiguous per row of BT
#pragma unroll
        for (int i = tid; i < 2048; i += 256) {
            int n = i >> 4, j = i & 15;
            ((u32*)&Btl[0][0])[i] =
                ((const u32*)BT)[(((size_t)(n0 + n) * K + kk) >> 1) + j];
        }
        __syncthreads();
        const u16* Bt = &Btl[0][0];
#endif

        // A fragment: two 16B contiguous runs from global
        Frag a;
        {
            const u32* ap = Ag + (kk >> 1);
            a.q[0] = *(const v4u*)(ap + hi * 4);
            a.q[1] = *(const v4u*)(ap + 8 + hi * 4);
        }

#pragma unroll
        for (int nb = 0; nb < 8; ++nb) {
            Frag b;
            const u32* bp = (const u32*)(Bt + (nb * 16 + lane16) * 32);
            b.q[0] = *(const v4u*)(bp + hi * 4);
            b.q[1] = *(const v4u*)(bp + 8 + hi * 4);
            acc[nb] = wmma_bf16(a.v, b.v, acc[nb]);
        }
        __syncthreads();
    }

#pragma unroll
    for (int nb = 0; nb < 8; ++nb) {
        int col = n0 + nb * 16 + lane16;
#pragma unroll
        for (int i = 0; i < 8; ++i) {
            int r = m0 + wave * 16 + i + 8 * hi;
            C[(size_t)r * N + col] = acc[nb][i];
        }
    }
}

// ---------------------------------------------------------------------------
// Flash attention (causal, GQA 16 q-heads / 4 kv-heads, hd=128).
// 128 threads = 4 waves; block = (head, 64-row q tile); key tiles of 64.
// Q in registers; K/V stream from L2 as b128 fragments; P staged through
// wave-private LDS only for the C-layout -> A-layout conversion.
// ---------------------------------------------------------------------------
__global__ __launch_bounds__(128)
void attn_kernel(const u16* __restrict__ Qb,   // (S,16,128) bf16, roped
                 const u16* __restrict__ Kb,   // (S, 4,128) bf16, roped
                 const u16* __restrict__ Vt,   // [4][128][2048] bf16
                 u16* __restrict__ ctx) {      // (S,16,128) bf16
    __shared__ u16 Pl[64 * 64];

    int tid  = threadIdx.x;
    int wave = tid >> 5;
    int lane = tid & 31;
    int lane16 = lane & 15;
    int hi     = lane >> 4;

    int h   = blockIdx.x;   // 0..15
    int qt  = blockIdx.y;   // 0..31
    int hkv = h >> 2;

    const float scale = 0.08838834764831845f;  // 1/sqrt(128)

    // --- Q fragments: 4 K-chunks of 32, two b128 loads each ---
    int qrow = qt * 64 + wave * 16 + lane16;
    const u32* Qg = (const u32*)(Qb + (((size_t)qrow * 16 + h) * 128));
    Frag qf[4];
#pragma unroll
    for (int c = 0; c < 4; ++c) {
        qf[c].q[0] = *(const v4u*)(Qg + c * 16 + hi * 4);
        qf[c].q[1] = *(const v4u*)(Qg + c * 16 + 8 + hi * 4);
    }

    v8f o[8];
    float m_i[8], l_i[8];
#pragma unroll
    for (int nb = 0; nb < 8; ++nb) o[nb] = (v8f){0,0,0,0,0,0,0,0};
#pragma unroll
    for (int i = 0; i < 8; ++i) { m_i[i] = -3.0e38f; l_i[i] = 0.0f; }

    for (int kt = 0; kt <= qt; ++kt) {
        int kb = kt * 64;

        // prefetch next key tile into L2/WGP$
        if (kt < qt) {
            const u16* nk = Kb + (((size_t)(kb + 64 + (tid & 63)) * 4 + hkv) * 128);
            __builtin_prefetch(nk, 0, 3);
        }

        // ---- S = Q @ K^T : 4 key sub-tiles of 16 ----
        v8f s[4];
#pragma unroll
        for (int nb = 0; nb < 4; ++nb) {
            v8f a = (v8f){0,0,0,0,0,0,0,0};
            int key = kb + nb * 16 + lane16;
            const u32* Kg = (const u32*)(Kb + (((size_t)key * 4 + hkv) * 128));
#pragma unroll
            for (int c = 0; c < 4; ++c) {
                Frag b;
                b.q[0] = *(const v4u*)(Kg + c * 16 + hi * 4);
                b.q[1] = *(const v4u*)(Kg + c * 16 + 8 + hi * 4);
                a = wmma_bf16(qf[c].v, b.v, a);
            }
            s[nb] = a;
        }

        // ---- scale + causal mask (registers) ----
        bool diag = (kt == qt);
#pragma unroll
        for (int nb = 0; nb < 4; ++nb)
#pragma unroll
            for (int i = 0; i < 8; ++i) {
                float v = s[nb][i] * scale;
                if (diag) {
                    int q = qt * 64 + wave * 16 + i + 8 * hi;
                    int k = kb + nb * 16 + lane16;
                    if (k > q) v = -3.0e38f;
                }
                s[nb][i] = v;
            }

        // ---- online softmax: row reductions inside 16-lane halves ----
        float mnew[8], alpha[8];
#pragma unroll
        for (int i = 0; i < 8; ++i) {
            float mx = m_i[i];
#pragma unroll
            for (int nb = 0; nb < 4; ++nb) {
                float v = s[nb][i];
#pragma unroll
                for (int off = 1; off < 16; off <<= 1)
                    v = fmaxf(v, __shfl_xor(v, off, 32));
                mx = fmaxf(mx, v);
            }
            mnew[i]  = mx;
            alpha[i] = __expf(m_i[i] - mx);
            m_i[i]   = mx;
        }
#pragma unroll
        for (int i = 0; i < 8; ++i) {
            float sum = 0.0f;
#pragma unroll
            for (int nb = 0; nb < 4; ++nb) {
                float p = __expf(s[nb][i] - mnew[i]);
                s[nb][i] = p;
#pragma unroll
                for (int off = 1; off < 16; off <<= 1)
                    p += __shfl_xor(p, off, 32);
                sum += p;
            }
            l_i[i] = l_i[i] * alpha[i] + sum;
        }

        // ---- stage P (bf16) into wave-private LDS rows ----
#pragma unroll
        for (int nb = 0; nb < 4; ++nb)
#pragma unroll
            for (int i = 0; i < 8; ++i) {
                int r = wave * 16 + i + 8 * hi;
                Pl[r * 64 + nb * 16 + lane16] = f2bf(s[nb][i]);
            }

        // ---- rescale running O ----
#pragma unroll
        for (int nb = 0; nb < 8; ++nb)
#pragma unroll
            for (int i = 0; i < 8; ++i)
                o[nb][i] *= alpha[i];

        // ---- O += P @ V ----
        Frag pa[2];
        {
            const u32* pp = (const u32*)&Pl[(wave * 16 + lane16) * 64];
#pragma unroll
            for (int c = 0; c < 2; ++c) {
                pa[c].q[0] = *(const v4u*)(pp + c * 16 + hi * 4);
                pa[c].q[1] = *(const v4u*)(pp + c * 16 + 8 + hi * 4);
            }
        }
#pragma unroll
        for (int nb = 0; nb < 8; ++nb) {
            int d = nb * 16 + lane16;
            const u32* Vg = (const u32*)(Vt + (((size_t)hkv * 128 + d) * 2048 + kb));
#pragma unroll
            for (int c = 0; c < 2; ++c) {
                Frag b;
                b.q[0] = *(const v4u*)(Vg + c * 16 + hi * 4);
                b.q[1] = *(const v4u*)(Vg + c * 16 + 8 + hi * 4);
                o[nb] = wmma_bf16(pa[c].v, b.v, o[nb]);
            }
        }
    }

    // ---- epilogue: O /= l, write bf16 ctx (S,16,128) ----
#pragma unroll
    for (int nb = 0; nb < 8; ++nb)
#pragma unroll
        for (int i = 0; i < 8; ++i) {
            int sq = qt * 64 + wave * 16 + i + 8 * hi;
            int d  = nb * 16 + lane16;
            float val = o[nb][i] / l_i[i];
            ctx[((size_t)sq * 16 + h) * 128 + d] = f2bf(val);
        }
}

// ---------------------------------------------------------------------------
// launcher
// ---------------------------------------------------------------------------
static inline size_t align256(size_t x) { return (x + 255) & ~(size_t)255; }

extern "C" void kernel_launch(void* const* d_in, const int* in_sizes, int n_in,
                              void* d_out, int out_size, void* d_ws, size_t ws_size,
                              hipStream_t stream) {
    (void)in_sizes; (void)n_in; (void)out_size; (void)ws_size;

    const float* X  = (const float*)d_in[0];   // (1,2048,2048)
    const float* Wq = (const float*)d_in[1];   // (2048,2048)
    const float* Wk = (const float*)d_in[2];   // (2048,512)
    const float* Wv = (const float*)d_in[3];   // (2048,512)
    const float* Wo = (const float*)d_in[4];   // (2048,2048)
    float* Out = (float*)d_out;                // (1,2048,2048)

    const int S = 2048, H = 2048, QD = 2048, KD = 512;

    char* ws = (char*)d_ws;
    size_t off = 0;
    auto carve = [&](size_t bytes) { char* p = ws + off; off += align256(bytes); return p; };

    u16*   xb  = (u16*)carve((size_t)S * H * 2);
    u16*   wqT = (u16*)carve((size_t)H * QD * 2);   // [QD][H]
    u16*   wkT = (u16*)carve((size_t)H * KD * 2);   // [KD][H]
    u16*   wvT = (u16*)carve((size_t)H * KD * 2);   // [KD][H]
    u16*   woT = (u16*)carve((size_t)QD * H * 2);   // [H][QD]
    float* Qf  = (float*)carve((size_t)S * QD * 4);
    float* Kf  = (float*)carve((size_t)S * KD * 4);
    float* Vf  = (float*)carve((size_t)S * KD * 4);
    u16*   Qb  = (u16*)carve((size_t)S * QD * 2);
    u16*   Kb  = (u16*)carve((size_t)S * KD * 2);
    u16*   Vt  = (u16*)carve((size_t)S * KD * 2);
    u16*   ctx = (u16*)carve((size_t)S * QD * 2);

    // 1) activation convert + weight transposes (fp32 -> bf16)
    cvt_bf16_kernel<<<(S * H + 255) / 256, 256, 0, stream>>>(X, xb, S * H);
    wtrans_kernel<<<(H * QD + 255) / 256, 256, 0, stream>>>(Wq, wqT, H, QD);
    wtrans_kernel<<<(H * KD + 255) / 256, 256, 0, stream>>>(Wk, wkT, H, KD);
    wtrans_kernel<<<(H * KD + 255) / 256, 256, 0, stream>>>(Wv, wvT, H, KD);
    wtrans_kernel<<<(QD * H + 255) / 256, 256, 0, stream>>>(Wo, woT, QD, H);

    // 2) QKV projections (bf16 WMMA GEMM, fp32 out)
    gemm_bf16_kernel<<<dim3(QD / 128, S / 128), 256, 0, stream>>>(xb, wqT, Qf, S, QD, H);
    gemm_bf16_kernel<<<dim3(KD / 128, S / 128), 256, 0, stream>>>(xb, wkT, Kf, S, KD, H);
    gemm_bf16_kernel<<<dim3(KD / 128, S / 128), 256, 0, stream>>>(xb, wvT, Vf, S, KD, H);

    // 3) RoPE (fp32 -> bf16) + V transpose
    {
        int nq = S * 16 * 64;
        rope_kernel<<<(nq + 255) / 256, 256, 0, stream>>>(Qf, Qb, 16);
        int nk = S * 4 * 64;
        rope_kernel<<<(nk + 255) / 256, 256, 0, stream>>>(Kf, Kb, 4);
        int nv = S * 4 * 128;
        vtrans_kernel<<<(nv + 255) / 256, 256, 0, stream>>>(Vf, Vt);
    }

    // 4) causal flash attention (bf16 WMMA, fp32 softmax state)
    attn_kernel<<<dim3(16, S / 64), 128, 0, stream>>>(Qb, Kb, Vt, ctx);

    // 5) output projection
    gemm_bf16_kernel<<<dim3(H / 128, S / 128), 256, 0, stream>>>(ctx, woT, Out, S, H, QD);
}
